// GravNet_Model_7292854469339
// MI455X (gfx1250) — matrix operand order repaired and checked by reading, not compiled, not run on hardware
//
#include <hip/hip_runtime.h>

#define NNODES 8192
#define SDIM 4
#define PDIM 22
#define KNN 16

typedef __attribute__((ext_vector_type(16))) _Float16 v16h;
typedef __attribute__((ext_vector_type(8)))  _Float16 v8h;
typedef __attribute__((ext_vector_type(8)))  float    v8f;

union V16U { v16h v; v8h h[2]; };

// gfx1250 hardware tanh (V_TANH_F32). Builtin if available, else inline asm.
// Trailing v_nop in the asm path covers the TRANS32 co-execution hazard
// (1 independent op required after a trans op before its output is consumed).
__device__ __forceinline__ float fast_tanh(float x) {
#if __has_builtin(__builtin_amdgcn_tanhf)
  return __builtin_amdgcn_tanhf(x);
#else
  float r;
  asm volatile("v_tanh_f32 %0, %1\n\tv_nop" : "=v"(r) : "v"(x));
  return r;
#endif
}

// One 16x16 C-tile accumulated over K=64 (two 16x16x32 WMMAs).
// act: [64 rows][64 K] f16 row-major in LDS (wave owns rows rowbase..rowbase+15)
// wT : [64 cols][64 K] f16 (weight transposed) in LDS; ctile selects 16 columns.
__device__ __forceinline__ v8f wmma_acc64(const _Float16* __restrict__ act,
                                          const _Float16* __restrict__ wT,
                                          int rowbase, int ctile, int lane, v8f acc) {
  const int lh = lane >> 4;        // lane group (0 or 1)
  const int ln = lane & 15;        // row (A) / column (B) within tile
  const _Float16* arow = act + (rowbase + ln) * 64;
  const _Float16* bcol = wT + (ctile * 16 + ln) * 64;
  #pragma unroll
  for (int kk = 0; kk < 64; kk += 32) {
    V16U a, b;
    // A 16x32 f16 layout: lanes<16: K 0..7 then 16..23 ; lanes>=16: K 8..15 then 24..31
    a.h[0] = *(const v8h*)(arow + kk + lh * 8);
    a.h[1] = *(const v8h*)(arow + kk + 16 + lh * 8);
    // B 32x16 f16 layout: lanes<16: K 0..7 then 16..23 ; lanes>=16: K 8..15 then 24..31
    b.h[0] = *(const v8h*)(bcol + kk + lh * 8);
    b.h[1] = *(const v8h*)(bcol + kk + 16 + lh * 8);
    acc = __builtin_amdgcn_wmma_f32_16x16x32_f16(false, a.v, false, b.v,
                                                 (short)0, acc, false, false);
  }
  return acc;
}

// Load W[K x ncols] (f32, row-major) transposed + zero-padded into sW[64 cols][64 K] f16.
__device__ __forceinline__ void load_wT(const float* __restrict__ W, int K, int ncols,
                                        _Float16* sW, int tid, int nthreads) {
  __syncthreads();
  for (int i = tid; i < 64 * 64; i += nthreads) sW[i] = (_Float16)0.f;
  __syncthreads();
  for (int i = tid; i < K * ncols; i += nthreads) {
    int k = i / ncols, n = i - k * ncols;
    sW[n * 64 + k] = (_Float16)W[i];
  }
  __syncthreads();
}

__device__ __forceinline__ void load_bias(const float* __restrict__ b, int nb,
                                          float* sBias, int tid) {
  if (tid < 64) sBias[tid] = (tid < nb) ? b[tid] : 0.f;
  __syncthreads();
}

// 64-row x 64-col GEMM tile: dst_f16 = act(tanh?)(src @ W + bias); optional f32 global copy.
__device__ __forceinline__ void gemm_store(const _Float16* __restrict__ src,
                                           const _Float16* __restrict__ sW,
                                           const float* __restrict__ sBias,
                                           _Float16* __restrict__ dst,
                                           float* __restrict__ gout, int row0,
                                           int rowbase, int lane, bool do_tanh) {
  const int lh = lane >> 4;
  #pragma unroll
  for (int c = 0; c < 4; ++c) {
    v8f acc = {};
    acc = wmma_acc64(src, sW, rowbase, c, lane, acc);
    const int col = c * 16 + (lane & 15);
    const float bb = sBias[col];
    #pragma unroll
    for (int r2 = 0; r2 < 8; ++r2) {
      float v = acc[r2] + bb;
      if (do_tanh) v = fast_tanh(v);
      const int rl = rowbase + r2 + lh * 8;
      dst[rl * 64 + col] = (_Float16)v;
      if (gout) gout[(row0 + rl) * 64 + col] = v;
    }
  }
}

// ---- Kernel 1: MLP (3 layers) + s/feat projections + |s|^2, all via WMMA ----
__global__ void __launch_bounds__(128)
mlp_block_kernel(const float* __restrict__ in, int in_dim,
                 const float* __restrict__ W1, const float* __restrict__ b1,
                 const float* __restrict__ W2, const float* __restrict__ b2,
                 const float* __restrict__ W3, const float* __restrict__ b3,
                 const float* __restrict__ Ws, const float* __restrict__ bs,
                 const float* __restrict__ Wh, const float* __restrict__ bh,
                 float* __restrict__ h_out, float* __restrict__ s_out,
                 float* __restrict__ sq_out, float* __restrict__ feat_out) {
  __shared__ _Float16 sA[64 * 64];
  __shared__ _Float16 sB[64 * 64];
  __shared__ _Float16 sW[64 * 64];
  __shared__ float sBias[64];
  __shared__ float sS[64 * SDIM];

  const int tid = threadIdx.x;
  const int lane = tid & 31;
  const int wave = tid >> 5;
  const int rowbase = wave * 16;
  const int row0 = blockIdx.x * 64;
  const int lh = lane >> 4;

  // stage input tile (zero-padded to K=64) as f16
  for (int i = tid; i < 64 * 64; i += 128) sA[i] = (_Float16)0.f;
  __syncthreads();
  for (int i = tid; i < 64 * in_dim; i += 128) {
    int r = i / in_dim, c = i - r * in_dim;
    sA[r * 64 + c] = (_Float16)in[(row0 + r) * in_dim + c];
  }
  // layer 1: sA -> sB (tanh)
  load_wT(W1, in_dim, 64, sW, tid, 128);
  load_bias(b1, 64, sBias, tid);
  gemm_store(sA, sW, sBias, sB, nullptr, row0, rowbase, lane, true);
  // layer 2: sB -> sA (tanh)
  load_wT(W2, 64, 64, sW, tid, 128);
  load_bias(b2, 64, sBias, tid);
  gemm_store(sB, sW, sBias, sA, nullptr, row0, rowbase, lane, true);
  // layer 3: sA -> sB (linear) + f32 h to global
  load_wT(W3, 64, 64, sW, tid, 128);
  load_bias(b3, 64, sBias, tid);
  gemm_store(sA, sW, sBias, sB, h_out, row0, rowbase, lane, false);

  // s = h @ Ws + bs  (pad 4 -> 16 cols, one WMMA tile)
  load_wT(Ws, 64, SDIM, sW, tid, 128);
  load_bias(bs, SDIM, sBias, tid);
  {
    v8f acc = {};
    acc = wmma_acc64(sB, sW, rowbase, 0, lane, acc);
    const int col = lane & 15;
    if (col < SDIM) {
      const float bb = sBias[col];
      #pragma unroll
      for (int r2 = 0; r2 < 8; ++r2) {
        const int rl = rowbase + r2 + lh * 8;
        const float v = acc[r2] + bb;
        sS[rl * SDIM + col] = v;
        s_out[(row0 + rl) * SDIM + col] = v;
      }
    }
  }
  __syncthreads();
  if (tid < 64) {
    float q = 0.f;
    #pragma unroll
    for (int c = 0; c < SDIM; ++c) { float v = sS[tid * SDIM + c]; q += v * v; }
    sq_out[row0 + tid] = q;
  }

  // feat = h @ Wh + bh  (pad 22 -> 32 cols, two WMMA tiles), stored [N x 32]
  load_wT(Wh, 64, PDIM, sW, tid, 128);
  load_bias(bh, PDIM, sBias, tid);
  #pragma unroll
  for (int c = 0; c < 2; ++c) {
    v8f acc = {};
    acc = wmma_acc64(sB, sW, rowbase, c, lane, acc);
    const int col = c * 16 + (lane & 15);
    const float bb = sBias[col];
    #pragma unroll
    for (int r2 = 0; r2 < 8; ++r2) {
      const int rl = rowbase + r2 + lh * 8;
      feat_out[(row0 + rl) * 32 + col] = acc[r2] + bb;
    }
  }
}

// ---- Kernel 2: brute-force kNN (top-16 by d^2) + weighted mean/max aggregation ----
__global__ void __launch_bounds__(256)
knn_agg_kernel(const float* __restrict__ s, const float* __restrict__ sq,
               const float* __restrict__ feat, float* __restrict__ agg) {
  __shared__ float4 ts[256];
  __shared__ float tsq[256];
  const int i = blockIdx.x * 256 + threadIdx.x;
  const float4 si = ((const float4*)s)[i];
  const float sqi = sq[i];
  float bd[KNN]; int bi[KNN];
  #pragma unroll
  for (int k = 0; k < KNN; ++k) { bd[k] = 3.4e38f; bi[k] = 0; }
  float worst = 3.4e38f; int wslot = 0;
  for (int t = 0; t < NNODES / 256; ++t) {
    __syncthreads();
    ts[threadIdx.x] = ((const float4*)s)[t * 256 + threadIdx.x];
    tsq[threadIdx.x] = sq[t * 256 + threadIdx.x];
    if (t + 1 < NNODES / 256) {
      // gfx1250 global_prefetch of the next coordinate tile (stream through L2)
      __builtin_prefetch(&((const float4*)s)[(t + 1) * 256 + threadIdx.x], 0, 1);
    }
    __syncthreads();
    for (int jj = 0; jj < 256; ++jj) {
      const float4 sv = ts[jj];
      float d2 = sqi + tsq[jj] -
                 2.f * (si.x * sv.x + si.y * sv.y + si.z * sv.z + si.w * sv.w);
      d2 = fmaxf(d2, 0.f);
      if (d2 < worst) {   // strict <: keeps lowest index on ties, matches top_k
        bd[wslot] = d2; bi[wslot] = t * 256 + jj;
        worst = bd[0]; wslot = 0;
        #pragma unroll
        for (int k = 1; k < KNN; ++k)
          if (bd[k] > worst) { worst = bd[k]; wslot = k; }
      }
    }
  }
  float mean[PDIM], mx[PDIM];
  #pragma unroll
  for (int c = 0; c < PDIM; ++c) { mean[c] = 0.f; mx[c] = -3.4e38f; }
  for (int k = 0; k < KNN; ++k) {
    const float w = __expf(-10.f * bd[k]);   // bare v_exp_f32 path
    const float* fr = feat + bi[k] * 32;
    #pragma unroll
    for (int c = 0; c < PDIM; ++c) {
      const float v = fr[c] * w;
      mean[c] += v;
      mx[c] = fmaxf(mx[c], v);
    }
  }
  float* ar = agg + i * 64;
  #pragma unroll
  for (int c = 0; c < PDIM; ++c) {
    ar[c] = mean[c] * (1.f / (float)KNN);
    ar[PDIM + c] = mx[c];
  }
  for (int c = 2 * PDIM; c < 64; ++c) ar[c] = 0.f;  // zero pad for WMMA K=64
}

// ---- Kernel 3: block output = h@Wo1 + agg@Wo2(pad) + bo2, dual WMMA chain ----
__global__ void __launch_bounds__(128)
gravnet_out_kernel(const float* __restrict__ h, const float* __restrict__ agg,
                   const float* __restrict__ Wo1, const float* __restrict__ Wo2,
                   const float* __restrict__ bo2, float* __restrict__ out) {
  __shared__ _Float16 sH[64 * 64];
  __shared__ _Float16 sG[64 * 64];
  __shared__ _Float16 sW1[64 * 64];
  __shared__ _Float16 sW2[64 * 64];
  __shared__ float sBias[64];
  const int tid = threadIdx.x, lane = tid & 31, wave = tid >> 5;
  const int rowbase = wave * 16, row0 = blockIdx.x * 64;
  const int lh = lane >> 4;
  for (int i = tid; i < 64 * 64; i += 128) {
    sH[i] = (_Float16)h[row0 * 64 + i];
    sG[i] = (_Float16)agg[row0 * 64 + i];
  }
  load_wT(Wo1, 64, 64, sW1, tid, 128);
  load_wT(Wo2, 2 * PDIM, 64, sW2, tid, 128);
  load_bias(bo2, 64, sBias, tid);
  #pragma unroll
  for (int c = 0; c < 4; ++c) {
    v8f acc = {};
    acc = wmma_acc64(sH, sW1, rowbase, c, lane, acc);
    acc = wmma_acc64(sG, sW2, rowbase, c, lane, acc);
    const int col = c * 16 + (lane & 15);
    const float bb = sBias[col];
    #pragma unroll
    for (int r2 = 0; r2 < 8; ++r2) {
      const int rl = rowbase + r2 + lh * 8;
      out[(row0 + rl) * 64 + col] = acc[r2] + bb;
    }
  }
}

// ---- Kernel 4: global max pool over N + 256->128 ReLU -> 128->2 ----
__global__ void __launch_bounds__(256)
final_kernel(const float* __restrict__ outs, const float* __restrict__ Wf1,
             const float* __restrict__ bf1, const float* __restrict__ Wf2,
             const float* __restrict__ bf2, float* __restrict__ dout) {
  __shared__ float pooled[256];
  __shared__ float hidden[128];
  const int t = threadIdx.x;
  const float* p = outs + (t >> 6) * (NNODES * 64) + (t & 63);
  float mx = -3.4e38f;
  for (int r = 0; r < NNODES; ++r) mx = fmaxf(mx, p[r * 64]);
  pooled[t] = mx;
  __syncthreads();
  if (t < 128) {
    float a = bf1[t];
    for (int k = 0; k < 256; ++k) a += pooled[k] * Wf1[k * 128 + t];
    hidden[t] = fmaxf(a, 0.f);
  }
  __syncthreads();
  if (t < 2) {
    float a = bf2[t];
    for (int k = 0; k < 128; ++k) a += hidden[k] * Wf2[k * 2 + t];
    dout[t] = a;
  }
}

extern "C" void kernel_launch(void* const* d_in, const int* in_sizes, int n_in,
                              void* d_out, int out_size, void* d_ws, size_t ws_size,
                              hipStream_t stream) {
  const float* x = (const float*)d_in[0];
  const float *Wf1, *bf1, *Wf2, *bf2;
  const float *W1[4], *b1[4], *W2[4], *b2[4], *W3[4], *b3[4];
  const float *Wsp[4], *bsp[4], *Whp[4], *bhp[4], *Wo1p[4], *Wo2p[4], *bo2p[4];

  if (n_in > 1 && in_sizes[1] == 256 * 128) {
    // jax tree_flatten order: sorted keys (W* before b*), None (Wo1 bias) dropped
    Wf1 = (const float*)d_in[1]; Wf2 = (const float*)d_in[2];
    bf1 = (const float*)d_in[3]; bf2 = (const float*)d_in[4];
    for (int b = 0; b < 4; ++b) {
      const int o = 5 + b * 13;
      W1[b]   = (const float*)d_in[o + 0];
      W2[b]   = (const float*)d_in[o + 1];
      W3[b]   = (const float*)d_in[o + 2];
      Whp[b]  = (const float*)d_in[o + 3];
      Wo1p[b] = (const float*)d_in[o + 4];
      Wo2p[b] = (const float*)d_in[o + 5];
      Wsp[b]  = (const float*)d_in[o + 6];
      b1[b]   = (const float*)d_in[o + 7];
      b2[b]   = (const float*)d_in[o + 8];
      b3[b]   = (const float*)d_in[o + 9];
      bhp[b]  = (const float*)d_in[o + 10];
      bo2p[b] = (const float*)d_in[o + 11];
      bsp[b]  = (const float*)d_in[o + 12];
    }
  } else {
    // dict insertion order: blocks first, then Wf1, bf1, Wf2, bf2
    for (int b = 0; b < 4; ++b) {
      const int o = 1 + b * 13;
      W1[b]  = (const float*)d_in[o + 0];  b1[b]   = (const float*)d_in[o + 1];
      W2[b]  = (const float*)d_in[o + 2];  b2[b]   = (const float*)d_in[o + 3];
      W3[b]  = (const float*)d_in[o + 4];  b3[b]   = (const float*)d_in[o + 5];
      Wsp[b] = (const float*)d_in[o + 6];  bsp[b]  = (const float*)d_in[o + 7];
      Whp[b] = (const float*)d_in[o + 8];  bhp[b]  = (const float*)d_in[o + 9];
      Wo1p[b] = (const float*)d_in[o + 10];
      Wo2p[b] = (const float*)d_in[o + 11];
      bo2p[b] = (const float*)d_in[o + 12];
    }
    Wf1 = (const float*)d_in[53]; bf1 = (const float*)d_in[54];
    Wf2 = (const float*)d_in[55]; bf2 = (const float*)d_in[56];
  }

  // Workspace layout (floats): h | s | sq | feat(pad32) | agg(pad64) | outs[4]
  float* ws = (float*)d_ws;
  float* hbuf    = ws;                                  // N*64
  float* sbuf    = hbuf + (size_t)NNODES * 64;          // N*4
  float* sqbuf   = sbuf + (size_t)NNODES * 4;           // N
  float* featbuf = sqbuf + (size_t)NNODES;              // N*32
  float* aggbuf  = featbuf + (size_t)NNODES * 32;       // N*64
  float* outsbuf = aggbuf + (size_t)NNODES * 64;        // 4 * N*64

  for (int b = 0; b < 4; ++b) {
    const float* in = (b == 0) ? x : (outsbuf + (size_t)(b - 1) * NNODES * 64);
    const int in_dim = (b == 0) ? 16 : 64;
    mlp_block_kernel<<<NNODES / 64, 128, 0, stream>>>(
        in, in_dim, W1[b], b1[b], W2[b], b2[b], W3[b], b3[b],
        Wsp[b], bsp[b], Whp[b], bhp[b], hbuf, sbuf, sqbuf, featbuf);
    knn_agg_kernel<<<NNODES / 256, 256, 0, stream>>>(sbuf, sqbuf, featbuf, aggbuf);
    gravnet_out_kernel<<<NNODES / 64, 128, 0, stream>>>(
        hbuf, aggbuf, Wo1p[b], Wo2p[b], bo2p[b],
        outsbuf + (size_t)b * NNODES * 64);
  }
  final_kernel<<<1, 256, 0, stream>>>(outsbuf, Wf1, bf1, Wf2, bf2, (float*)d_out);
}
